// MultiWayTransformerLayer_30219389894936
// MI455X (gfx1250) — matrix operand order repaired and checked by reading, not compiled, not run on hardware
//
#include <hip/hip_runtime.h>
#include <hip/hip_bf16.h>
#include <cstdint>

#define B_    8
#define S_    1024
#define H_    1024
#define NHEAD 16
#define DH_   64
#define DFF_  4096
#define NTOK  (B_ * S_)   // 8192

typedef __attribute__((ext_vector_type(16))) __bf16 bf16x16;
typedef __attribute__((ext_vector_type(8)))  float  f32x8;

union FragU { bf16x16 v; uint4 q[2]; };

static __device__ __forceinline__ f32x8 wmma_bf16(bf16x16 a, bf16x16 b, f32x8 c) {
  return __builtin_amdgcn_wmma_f32_16x16x32_bf16(false, a, false, b, (short)0, c, false, false);
}

// A-frag (16x32 bf16): lanes 0-15 row m, K {0-15}; lanes 16-31 row m, K {16-31}
static __device__ __forceinline__ bf16x16 load_afrag(const __bf16* base, int stride, int lane) {
  const __bf16* p = base + (lane & 15) * stride + ((lane & 16) ? 16 : 0);
  FragU f;
  f.q[0] = *(const uint4*)(p);
  f.q[1] = *(const uint4*)(p + 8);
  return f.v;
}
// B-frag (32x16 bf16, stored as B^T rows [N][K]): lanes 0-15 col n K{0-15}; lanes 16-31 K{16-31}
static __device__ __forceinline__ bf16x16 load_bfrag(const __bf16* base, int stride, int lane) {
  const __bf16* p = base + (lane & 15) * stride + ((lane & 16) ? 16 : 0);
  FragU f;
  f.q[0] = *(const uint4*)(p);
  f.q[1] = *(const uint4*)(p + 8);
  return f.v;
}

// ---------------- conversion kernels ----------------
__global__ __launch_bounds__(256) void cvt_kernel(const float* __restrict__ src,
                                                  __bf16* __restrict__ dst, int n) {
  for (int i = blockIdx.x * 256 + threadIdx.x; i < n; i += gridDim.x * 256)
    dst[i] = (__bf16)src[i];
}

// src [R,C] f32 -> dst [C,R] bf16
__global__ __launch_bounds__(256) void transpose_cvt_kernel(const float* __restrict__ src,
                                                            __bf16* __restrict__ dst,
                                                            int R, int C) {
  __shared__ float tile[32][33];
  int c0 = blockIdx.x * 32, r0 = blockIdx.y * 32;
#pragma unroll
  for (int j = 0; j < 4; ++j) {
    int r = r0 + threadIdx.y + j * 8;
    tile[threadIdx.y + j * 8][threadIdx.x] = src[(size_t)r * C + c0 + threadIdx.x];
  }
  __syncthreads();
#pragma unroll
  for (int j = 0; j < 4; ++j) {
    int c = c0 + threadIdx.y + j * 8;
    dst[(size_t)c * R + r0 + threadIdx.x] = (__bf16)tile[threadIdx.x][threadIdx.y + j * 8];
  }
}

// ---------------- MoE routing (meta: [0..2]=counts [3..5]=cursor [6..9]=offsets) ----------------
__global__ void route_init_kernel(int* meta) { if (threadIdx.x < 16) meta[threadIdx.x] = 0; }

__global__ __launch_bounds__(256) void route_count_kernel(const int* __restrict__ mm, int* meta) {
  int t = blockIdx.x * 256 + threadIdx.x;
  int m = mm[t]; m = m < 0 ? 0 : (m > 2 ? 2 : m);
  atomicAdd(&meta[m], 1);
}
__global__ void route_scan_kernel(int* meta) {
  meta[6] = 0; meta[7] = meta[0]; meta[8] = meta[0] + meta[1];
  meta[9] = meta[0] + meta[1] + meta[2];
}
__global__ __launch_bounds__(256) void route_scatter_kernel(const int* __restrict__ mm,
                                                            int* meta, int* perm) {
  int t = blockIdx.x * 256 + threadIdx.x;
  int m = mm[t]; m = m < 0 ? 0 : (m > 2 ? 2 : m);
  int pos = atomicAdd(&meta[3 + m], 1);
  perm[meta[6 + m] + pos] = t;
}

// ---------------- generic bf16 WMMA GEMM ----------------
// C[M,N] = A[M,K](bf16,row-major) * B^T  with B given as [N,K] bf16 row-major, + bias.
// MODE 0: store bf16 (QKV)
// MODE 1: += resid (f32), store f32 (out-proj)
// MODE 2: gathered A rows via perm, GELU, store bf16 compact rows (fc1)
// MODE 3: compact A rows, scatter to orig rows with += resid, store f32 (fc2)
#define BM  128
#define BN  64
#define BKK 64
#define LDT 72   // BKK + 8 pad -> 144B row stride, 16B aligned chunks

template<int MODE>
__global__ __launch_bounds__(256) void gemm_kernel(
    const __bf16* __restrict__ A, int lda,
    const __bf16* __restrict__ B0, const __bf16* __restrict__ B1, const __bf16* __restrict__ B2,
    int K,
    const float* __restrict__ bias0, const float* __restrict__ bias1, const float* __restrict__ bias2,
    float* __restrict__ outF, __bf16* __restrict__ outB, int ldc,
    const float* __restrict__ resid,
    const int* __restrict__ meta, const int* __restrict__ perm)
{
  __shared__ __bf16 As[BM * LDT];
  __shared__ __bf16 Bs[BN * LDT];

  const int tid  = threadIdx.x;
  const int lane = tid & 31;
  const int wid  = tid >> 5;
  const int wm   = wid & 3;   // wave row block (32 rows)
  const int wn   = wid >> 2;  // wave col block (32 cols)
  const int mtile = blockIdx.x;
  const int ntile = blockIdx.y;
  const int z     = blockIdx.z;

  const __bf16* Bw = (z == 0) ? B0 : (z == 1) ? B1 : B2;
  const float*  bw = (z == 0) ? bias0 : (z == 1) ? bias1 : bias2;

  int cnt = NTOK, base = 0;
  if (MODE == 2 || MODE == 3) {
    cnt  = meta[z];
    base = meta[6 + z];
    if (mtile * BM >= cnt) return;
  }

  int arow[4];
#pragma unroll
  for (int i = 0; i < 4; ++i) {
    int r    = (tid >> 3) + i * 32;
    int rloc = mtile * BM + r;
    if (MODE == 2)      arow[i] = perm[base + ((rloc < cnt) ? rloc : 0)];
    else if (MODE == 3) arow[i] = base + ((rloc < cnt) ? rloc : 0);
    else                arow[i] = rloc;
  }
  const int kcA      = (tid & 7) * 8;
  const int browBase = ntile * BN;

  f32x8 acc[2][2];
#pragma unroll
  for (int a = 0; a < 2; ++a)
#pragma unroll
    for (int b = 0; b < 2; ++b) acc[a][b] = {};

  for (int k0 = 0; k0 < K; k0 += BKK) {
#pragma unroll
    for (int i = 0; i < 4; ++i) {
      int r = (tid >> 3) + i * 32;
      uint4 d = *(const uint4*)(A + (size_t)arow[i] * lda + k0 + kcA);
      *(uint4*)(&As[r * LDT + kcA]) = d;
    }
#pragma unroll
    for (int i = 0; i < 2; ++i) {
      int r = (tid >> 3) + i * 32;
      uint4 d = *(const uint4*)(Bw + (size_t)(browBase + r) * K + k0 + kcA);
      *(uint4*)(&Bs[r * LDT + kcA]) = d;
    }
    if (k0 + BKK < K) {
      __builtin_prefetch(A + (size_t)arow[0] * lda + k0 + BKK + kcA, 0, 1);
      __builtin_prefetch(Bw + (size_t)(browBase + (tid >> 3)) * K + k0 + BKK + kcA, 0, 1);
    }
    __syncthreads();
#pragma unroll
    for (int s = 0; s < 2; ++s) {
      bf16x16 a0 = load_afrag(&As[(wm * 32 +  0) * LDT + s * 32], LDT, lane);
      bf16x16 a1 = load_afrag(&As[(wm * 32 + 16) * LDT + s * 32], LDT, lane);
      bf16x16 b0 = load_bfrag(&Bs[(wn * 32 +  0) * LDT + s * 32], LDT, lane);
      bf16x16 b1 = load_bfrag(&Bs[(wn * 32 + 16) * LDT + s * 32], LDT, lane);
      acc[0][0] = wmma_bf16(a0, b0, acc[0][0]);
      acc[0][1] = wmma_bf16(a0, b1, acc[0][1]);
      acc[1][0] = wmma_bf16(a1, b0, acc[1][0]);
      acc[1][1] = wmma_bf16(a1, b1, acc[1][1]);
    }
    __syncthreads();
  }

  const int nlo = lane & 15;
  const int rhi = (lane & 16) ? 8 : 0;
#pragma unroll
  for (int tm = 0; tm < 2; ++tm) {
#pragma unroll
    for (int tn = 0; tn < 2; ++tn) {
      int col  = ntile * BN + wn * 32 + tn * 16 + nlo;
      float bv = bw[col];
#pragma unroll
      for (int r = 0; r < 8; ++r) {
        int rloc = mtile * BM + wm * 32 + tm * 16 + r + rhi;
        float v  = acc[tm][tn][r] + bv;
        if (MODE == 0) {
          outB[(size_t)rloc * ldc + col] = (__bf16)v;
        } else if (MODE == 1) {
          v += resid[(size_t)rloc * H_ + col];
          outF[(size_t)rloc * ldc + col] = v;
        } else if (MODE == 2) {
          if (rloc < cnt) {
            float g = 0.5f * v * (1.0f + erff(v * 0.70710678118654752f));
            outB[(size_t)(base + rloc) * ldc + col] = (__bf16)g;
          }
        } else {  // MODE 3
          if (rloc < cnt) {
            int orig = perm[base + rloc];
            outF[(size_t)orig * ldc + col] = v + resid[(size_t)orig * H_ + col];
          }
        }
      }
    }
  }
}

// ---------------- attention: block = (b, 16-query tile), wave = head ----------------
#define VT_BYTES (1024 * 40 * 2)        // V^T staged [d=1024][k=32] pad->40
#define PF_BYTES (16 * 16 * 32 * 4)     // per-head probs f32 [h][q][k]
#define PB_BYTES (16 * 16 * 40 * 2)     // per-head probs bf16 (A-frag staging, pad->40)
#define ATT_SMEM (VT_BYTES + PF_BYTES + PB_BYTES)  // 135168 B

__global__ __launch_bounds__(512) void attn_kernel(
    const __bf16* __restrict__ qkv, const int* __restrict__ amask,
    __bf16* __restrict__ ctxb, float* __restrict__ attnW)
{
  extern __shared__ char smem[];
  __bf16* Vt = (__bf16*)smem;
  float*  pF = (float*)(smem + VT_BYTES);
  __bf16* pB = (__bf16*)(smem + VT_BYTES + PF_BYTES);

  const int tid  = threadIdx.x;
  const int lane = tid & 31;
  const int h    = tid >> 5;                 // head
  const int b    = blockIdx.x >> 6;
  const int q0   = (blockIdx.x & 63) << 4;
  const int nlo  = lane & 15;
  const bool hiHalf = (lane & 16) != 0;

  // persistent Q fragments (direct global loads in WMMA A layout)
  bf16x16 qf[2];
  {
    const __bf16* qb = qkv + ((size_t)(b * S_ + q0 + nlo)) * (3 * H_) + h * DH_;
#pragma unroll
    for (int s = 0; s < 2; ++s) {
      FragU f;
      const __bf16* p = qb + s * 32 + (hiHalf ? 16 : 0);
      f.q[0] = *(const uint4*)(p);
      f.q[1] = *(const uint4*)(p + 8);
      qf[s] = f.v;
    }
  }
  const int* am = amask + b * S_;

  // ---- pass 1a: per-lane running max (no cross-lane traffic in loop) ----
  float m8[8];
#pragma unroll
  for (int r = 0; r < 8; ++r) m8[r] = -1e30f;

  for (int kt = 0; kt < S_ / 16; ++kt) {
    const __bf16* kb = qkv + ((size_t)(b * S_ + kt * 16 + nlo)) * (3 * H_) + H_ + h * DH_;
    f32x8 acc = {};
#pragma unroll
    for (int s = 0; s < 2; ++s) {
      FragU f;
      const __bf16* p = kb + s * 32 + (hiHalf ? 16 : 0);
      f.q[0] = *(const uint4*)(p);
      f.q[1] = *(const uint4*)(p + 8);
      acc = wmma_bf16(qf[s], f.v, acc);
    }
    float mv = am[kt * 16 + nlo] ? 0.0f : -1e30f;
#pragma unroll
    for (int r = 0; r < 8; ++r)
      m8[r] = fmaxf(m8[r], acc[r] * 0.125f + mv);
  }
  // single cross-lane butterfly per accumulator (row max, replicated to all lanes)
#pragma unroll
  for (int r = 0; r < 8; ++r) {
    m8[r] = fmaxf(m8[r], __shfl_xor(m8[r], 1));
    m8[r] = fmaxf(m8[r], __shfl_xor(m8[r], 2));
    m8[r] = fmaxf(m8[r], __shfl_xor(m8[r], 4));
    m8[r] = fmaxf(m8[r], __shfl_xor(m8[r], 8));
  }

  // ---- pass 1b: per-lane sum of exp(s - m) ----
  float l8[8];
#pragma unroll
  for (int r = 0; r < 8; ++r) l8[r] = 0.0f;

  for (int kt = 0; kt < S_ / 16; ++kt) {
    const __bf16* kb = qkv + ((size_t)(b * S_ + kt * 16 + nlo)) * (3 * H_) + H_ + h * DH_;
    f32x8 acc = {};
#pragma unroll
    for (int s = 0; s < 2; ++s) {
      FragU f;
      const __bf16* p = kb + s * 32 + (hiHalf ? 16 : 0);
      f.q[0] = *(const uint4*)(p);
      f.q[1] = *(const uint4*)(p + 8);
      acc = wmma_bf16(qf[s], f.v, acc);
    }
    float mv = am[kt * 16 + nlo] ? 0.0f : -1e30f;
#pragma unroll
    for (int r = 0; r < 8; ++r)
      l8[r] += __expf(acc[r] * 0.125f + mv - m8[r]);
  }
#pragma unroll
  for (int r = 0; r < 8; ++r) {
    l8[r] += __shfl_xor(l8[r], 1);
    l8[r] += __shfl_xor(l8[r], 2);
    l8[r] += __shfl_xor(l8[r], 4);
    l8[r] += __shfl_xor(l8[r], 8);
  }
  float rl8[8];
#pragma unroll
  for (int r = 0; r < 8; ++r) rl8[r] = 1.0f / l8[r];

  // ---- pass 2: probabilities, P*V, head-averaged attn output ----
  f32x8 O[4];
#pragma unroll
  for (int d = 0; d < 4; ++d) O[d] = {};

  const int vrow = tid & 31;
  const int vd0  = (tid >> 5) * 64;

  for (int ch = 0; ch < S_ / 32; ++ch) {
    const int kc = ch * 32;
    __syncthreads();
    {  // cooperative V^T staging: 32 keys x 1024 dims
      const __bf16* vsrc = qkv + ((size_t)(b * S_ + kc + vrow)) * (3 * H_) + 2 * H_ + vd0;
      union { uint4 q[8]; __bf16 e[64]; } blob;
#pragma unroll
      for (int i = 0; i < 8; ++i) blob.q[i] = *(const uint4*)(vsrc + i * 8);
#pragma unroll
      for (int j = 0; j < 64; ++j) Vt[(vd0 + j) * 40 + vrow] = blob.e[j];
    }
    __syncthreads();

#pragma unroll
    for (int t = 0; t < 2; ++t) {
      const int kb16 = kc + t * 16;
      const __bf16* kb = qkv + ((size_t)(b * S_ + kb16 + nlo)) * (3 * H_) + H_ + h * DH_;
      f32x8 acc = {};
#pragma unroll
      for (int s = 0; s < 2; ++s) {
        FragU f;
        const __bf16* p = kb + s * 32 + (hiHalf ? 16 : 0);
        f.q[0] = *(const uint4*)(p);
        f.q[1] = *(const uint4*)(p + 8);
        acc = wmma_bf16(qf[s], f.v, acc);
      }
      float mv = am[kb16 + nlo] ? 0.0f : -1e30f;
#pragma unroll
      for (int r = 0; r < 8; ++r) {
        int qq   = r + (hiHalf ? 8 : 0);
        float pr = __expf(acc[r] * 0.125f + mv - m8[r]) * rl8[r];
        pF[(h * 16 + qq) * 32 + t * 16 + nlo] = pr;
        pB[(h * 16 + qq) * 40 + t * 16 + nlo] = (__bf16)pr;
      }
    }
    // same-wave LDS store -> fragment load hazard
    asm volatile("s_wait_dscnt 0" ::: "memory");

    bf16x16 pfrag;
    {
      const __bf16* p = pB + (h * 16) * 40 + nlo * 40 + (hiHalf ? 16 : 0);
      FragU f;
      f.q[0] = *(const uint4*)(p);
      f.q[1] = *(const uint4*)(p + 8);
      pfrag = f.v;
    }
#pragma unroll
    for (int dt = 0; dt < 4; ++dt) {
      const __bf16* p = Vt + (h * 64 + dt * 16 + nlo) * 40 + (hiHalf ? 16 : 0);
      FragU f;
      f.q[0] = *(const uint4*)(p);
      f.q[1] = *(const uint4*)(p + 8);
      O[dt] = wmma_bf16(pfrag, f.v, O[dt]);
    }
    __syncthreads();
    {  // deterministic head average -> attn_weights
      int qq = tid >> 5;
      int kk = tid & 31;
      float ssum = 0.0f;
#pragma unroll
      for (int hh = 0; hh < 16; ++hh) ssum += pF[(hh * 16 + qq) * 32 + kk];
      attnW[((size_t)(b * S_ + q0 + qq)) * S_ + kc + kk] = ssum * 0.0625f;
    }
  }

  // write context (bf16) for the out-projection GEMM
#pragma unroll
  for (int dt = 0; dt < 4; ++dt)
#pragma unroll
    for (int r = 0; r < 8; ++r) {
      int qq = r + (hiHalf ? 8 : 0);
      ctxb[((size_t)(b * S_ + q0 + qq)) * H_ + h * DH_ + dt * 16 + nlo] = (__bf16)O[dt][r];
    }
}

// ---------------- LayerNorm (row per block) ----------------
__global__ __launch_bounds__(256) void ln_kernel(const float* src, const float* sc,
                                                 const float* bi, float* outF, __bf16* outB)
{
  __shared__ float redS[8], redQ[8], stats[2];
  const int tid  = threadIdx.x;
  const int lane = tid & 31;
  const int wid  = tid >> 5;
  const size_t row = blockIdx.x;
  const float* p = src + row * H_;
  float v[4], s = 0.f, q = 0.f;
#pragma unroll
  for (int j = 0; j < 4; ++j) {
    v[j] = p[tid + j * 256];
    s += v[j];
    q += v[j] * v[j];
  }
#pragma unroll
  for (int off = 16; off >= 1; off >>= 1) {
    s += __shfl_xor(s, off);
    q += __shfl_xor(q, off);
  }
  if (lane == 0) { redS[wid] = s; redQ[wid] = q; }
  __syncthreads();
  if (tid == 0) {
    float ts = 0.f, tq = 0.f;
    for (int i = 0; i < 8; ++i) { ts += redS[i]; tq += redQ[i]; }
    float mean = ts * (1.0f / H_);
    float var  = tq * (1.0f / H_) - mean * mean;
    stats[0] = mean;
    stats[1] = rsqrtf(var + 1e-5f);
  }
  __syncthreads();
  float mean = stats[0], rstd = stats[1];
#pragma unroll
  for (int j = 0; j < 4; ++j) {
    int c = tid + j * 256;
    float y = (v[j] - mean) * rstd * sc[c] + bi[c];
    outF[row * H_ + c] = y;
    if (outB) outB[row * H_ + c] = (__bf16)y;
  }
}

// ---------------- launch ----------------
extern "C" void kernel_launch(void* const* d_in, const int* in_sizes, int n_in,
                              void* d_out, int out_size, void* d_ws, size_t ws_size,
                              hipStream_t stream)
{
  const float* x         = (const float*)d_in[0];
  const int*   modality  = (const int*)d_in[1];
  const int*   amask     = (const int*)d_in[2];
  const float* in_proj_w = (const float*)d_in[3];
  const float* in_proj_b = (const float*)d_in[4];
  const float* out_w     = (const float*)d_in[5];
  const float* out_b     = (const float*)d_in[6];
  const float* ln1_s = (const float*)d_in[7],  *ln1_b = (const float*)d_in[8];
  const float* ln2_s = (const float*)d_in[9],  *ln2_b = (const float*)d_in[10];
  const float* w1[3] = { (const float*)d_in[11], (const float*)d_in[15], (const float*)d_in[19] };
  const float* b1[3] = { (const float*)d_in[12], (const float*)d_in[16], (const float*)d_in[20] };
  const float* w2[3] = { (const float*)d_in[13], (const float*)d_in[17], (const float*)d_in[21] };
  const float* b2[3] = { (const float*)d_in[14], (const float*)d_in[18], (const float*)d_in[22] };

  char* ws = (char*)d_ws;
  size_t off = 0;
  auto take = [&](size_t bytes) -> char* {
    char* p = ws + off;
    off += (bytes + 255) & ~(size_t)255;
    return p;
  };
  __bf16* xb    = (__bf16*)take((size_t)NTOK * H_ * 2);
  __bf16* wqkvb = (__bf16*)take((size_t)3 * H_ * H_ * 2);
  __bf16* owb   = (__bf16*)take((size_t)H_ * H_ * 2);
  __bf16* w1t[3]; for (int e = 0; e < 3; ++e) w1t[e] = (__bf16*)take((size_t)DFF_ * H_ * 2);
  __bf16* w2t[3]; for (int e = 0; e < 3; ++e) w2t[e] = (__bf16*)take((size_t)H_ * DFF_ * 2);
  __bf16* qkvb  = (__bf16*)take((size_t)NTOK * 3 * H_ * 2);
  __bf16* ctxb  = (__bf16*)take((size_t)NTOK * H_ * 2);
  float*  hpre  = (float*)take((size_t)NTOK * H_ * 4);   // x+attn, then h (in place)
  __bf16* hb    = (__bf16*)take((size_t)NTOK * H_ * 2);
  __bf16* t1    = (__bf16*)take((size_t)NTOK * DFF_ * 2);
  float*  hpre2 = (float*)take((size_t)NTOK * H_ * 4);
  int*    meta  = (int*)take(64 * 4);
  int*    perm  = (int*)take((size_t)NTOK * 4);

  float* outMain = (float*)d_out;
  float* outAttn = outMain + (size_t)NTOK * H_;

  // bf16 conversions / weight pre-transposition
  cvt_kernel<<<dim3(2048), dim3(256), 0, stream>>>(x, xb, NTOK * H_);
  cvt_kernel<<<dim3(768),  dim3(256), 0, stream>>>(in_proj_w, wqkvb, 3 * H_ * H_);
  cvt_kernel<<<dim3(256),  dim3(256), 0, stream>>>(out_w, owb, H_ * H_);
  for (int e = 0; e < 3; ++e) {
    transpose_cvt_kernel<<<dim3(DFF_ / 32, H_ / 32), dim3(32, 8), 0, stream>>>(w1[e], w1t[e], H_, DFF_);
    transpose_cvt_kernel<<<dim3(H_ / 32, DFF_ / 32), dim3(32, 8), 0, stream>>>(w2[e], w2t[e], DFF_, H_);
  }
  // routing
  route_init_kernel<<<1, 32, 0, stream>>>(meta);
  route_count_kernel<<<NTOK / 256, 256, 0, stream>>>(modality, meta);
  route_scan_kernel<<<1, 1, 0, stream>>>(meta);
  route_scatter_kernel<<<NTOK / 256, 256, 0, stream>>>(modality, meta, perm);

  // QKV projection
  gemm_kernel<0><<<dim3(NTOK / BM, (3 * H_) / BN, 1), 256, 0, stream>>>(
      xb, H_, wqkvb, wqkvb, wqkvb, H_, in_proj_b, in_proj_b, in_proj_b,
      nullptr, qkvb, 3 * H_, nullptr, nullptr, nullptr);

  // attention
  (void)hipFuncSetAttribute((const void*)attn_kernel,
                            hipFuncAttributeMaxDynamicSharedMemorySize, ATT_SMEM);
  attn_kernel<<<dim3(B_ * (S_ / 16)), dim3(512), ATT_SMEM, stream>>>(qkvb, amask, ctxb, outAttn);

  // out projection + residual
  gemm_kernel<1><<<dim3(NTOK / BM, H_ / BN, 1), 256, 0, stream>>>(
      ctxb, H_, owb, owb, owb, H_, out_b, out_b, out_b,
      hpre, nullptr, H_, x, nullptr, nullptr);

  // LayerNorm 1 (in place, plus bf16 copy)
  ln_kernel<<<dim3(NTOK), 256, 0, stream>>>(hpre, ln1_s, ln1_b, hpre, hb);

  // expert fc1 (gathered rows, GELU)
  gemm_kernel<2><<<dim3(NTOK / BM, DFF_ / BN, 3), 256, 0, stream>>>(
      hb, H_, w1t[0], w1t[1], w1t[2], H_, b1[0], b1[1], b1[2],
      nullptr, t1, DFF_, nullptr, meta, perm);

  // expert fc2 (scatter back + residual h)
  gemm_kernel<3><<<dim3(NTOK / BM, H_ / BN, 3), 256, 0, stream>>>(
      t1, DFF_, w2t[0], w2t[1], w2t[2], DFF_, b2[0], b2[1], b2[2],
      hpre2, nullptr, H_, hpre, meta, perm);

  // LayerNorm 2 -> main output
  ln_kernel<<<dim3(NTOK), 256, 0, stream>>>(hpre2, ln2_s, ln2_b, outMain, nullptr);

  (void)in_sizes; (void)n_in; (void)out_size; (void)ws_size;
}
